// ChebClassifier_69999376990332
// MI455X (gfx1250) — matrix operand order repaired and checked by reading, not compiled, not run on hardware
//
#include <hip/hip_runtime.h>
#include <hip/hip_bf16.h>

// Problem constants (match reference)
#define N_NODES   100000
#define N_EDGES   1600000
#define N_GRAPHS  50
#define DIM       128        // IN == HID
#define KDIM      256        // K*IN == K*HID
#define NCLS      10
#define ROWBLKS   (N_NODES / 16)   // 6250 (N divisible by 16)

typedef __attribute__((ext_vector_type(16))) __bf16 v16bf;
typedef __attribute__((ext_vector_type(8)))  float  v8f;
typedef __attribute__((ext_vector_type(4)))  int    v4i;

typedef __attribute__((address_space(1))) v4i as1_v4i;   // global
typedef __attribute__((address_space(3))) v4i as3_v4i;   // LDS

__device__ __forceinline__ unsigned short f2bf(float x) {
    // round-to-nearest-even f32 -> bf16
    unsigned u = __float_as_uint(x);
    unsigned r = u + 0x7FFFu + ((u >> 16) & 1u);
    return (unsigned short)(r >> 16);
}

// ---------------- utility: zero a float buffer ----------------
__global__ void zero_f32(float* __restrict__ p, long n) {
    long i = (long)blockIdx.x * blockDim.x + threadIdx.x;
    long stride = (long)gridDim.x * blockDim.x;
    for (; i < n; i += stride) p[i] = 0.0f;
}

// ---------------- degree + dinv (in place) ----------------
__global__ void degree_kernel(const int* __restrict__ dst, float* __restrict__ deg) {
    int e = blockIdx.x * blockDim.x + threadIdx.x;
    if (e < N_EDGES) atomicAdd(&deg[dst[e]], 1.0f);
}

__global__ void dinv_kernel(float* __restrict__ deg_dinv) {
    int i = blockIdx.x * blockDim.x + threadIdx.x;
    if (i < N_NODES) {
        float d = deg_dinv[i];
        d = d < 1.0f ? 1.0f : d;
        deg_dinv[i] = rsqrtf(d);
    }
}

// ---------------- SpMM: agg[dst] += feat[src] * dinv[src] ----------------
// 128 threads per edge (one per feature); 2 edges per 256-thread block.
__global__ void spmm_kernel(const int* __restrict__ src, const int* __restrict__ dst,
                            const float* __restrict__ feat, const float* __restrict__ dinv,
                            float* __restrict__ agg) {
    int idx = blockIdx.x * blockDim.x + threadIdx.x;
    int e = idx >> 7;
    int f = idx & 127;
    if (e < N_EDGES) {
        int s = src[e];
        int d = dst[e];
        float v = feat[(long)s * DIM + f] * dinv[s];
        atomicAdd(&agg[(long)d * DIM + f], v);
    }
}

// ---------------- pack activations into WMMA A-fragment order (bf16) -------
// Logical A row m has K=256: k<128 -> feat[m][k];  k>=128 -> -agg[m][k-128]*dinv[m].
// Fragment layout (16-bit A 16x32, ISA 7.12.2): lane l in 0..31 of a wave holds
// row m = l&15; dword v holds K pair base = (v<4?0:16) + (l>>4)*8 + (v&3)*2.
// Stored fragment-major: xafrag[((rowblk*8 + kt)*32 + lane)*16 + slot].
// Also re-zeroes agg for the next layer's accumulation.
__global__ void pack_a_kernel(const float* __restrict__ feat, float* __restrict__ agg,
                              const float* __restrict__ dinv,
                              unsigned short* __restrict__ xafrag) {
    int rowblk = blockIdx.x;            // 0..ROWBLKS-1
    int kt   = threadIdx.x >> 5;        // 0..7  (K tile of 32)
    int lane = threadIdx.x & 31;
    int m = rowblk * 16 + (lane & 15);
    float di = dinv[m];

    unsigned short tmp[16];
#pragma unroll
    for (int e = 0; e < 16; ++e) {
        int v = e >> 1;
        int k = kt * 32 + ((v & 4) ? 16 : 0) + ((lane >> 4) << 3) + ((v & 3) << 1) + (e & 1);
        float val;
        if (k < DIM) {
            val = feat[(long)m * DIM + k];
        } else {
            long idx = (long)m * DIM + (k - DIM);
            val = -agg[idx] * di;
            agg[idx] = 0.0f;            // reset for next accumulation pass
        }
        tmp[e] = f2bf(val);
    }
    unsigned short* dp = xafrag + (((long)rowblk * 8 + kt) * 32 + lane) * 16;
    ((uint4*)dp)[0] = *(const uint4*)&tmp[0];
    ((uint4*)dp)[1] = *(const uint4*)&tmp[8];
}

// ---------------- pack weights into WMMA B-fragment order (bf16) ----------
// B tile is 32x16 (KxN): lane l holds column n = l&15; element e holds K =
// kt*32 + (e<8?0:16) + (l>>4)*8 + (e&7) (same order as A). W is row-major (KDIM x DIM).
__global__ void pack_w_kernel(const float* __restrict__ W, unsigned short* __restrict__ wfrag) {
    int ntile = blockIdx.x;             // 0..7
    int kt   = threadIdx.x >> 5;        // 0..7
    int lane = threadIdx.x & 31;
    int n = ntile * 16 + (lane & 15);

    unsigned short tmp[16];
#pragma unroll
    for (int e = 0; e < 16; ++e) {
        int k = kt * 32 + ((e & 8) ? 16 : 0) + ((lane >> 4) << 3) + (e & 7);
        tmp[e] = f2bf(W[(long)k * DIM + n]);
    }
    unsigned short* dp = wfrag + (((long)ntile * 8 + kt) * 32 + lane) * 16;
    ((uint4*)dp)[0] = *(const uint4*)&tmp[0];
    ((uint4*)dp)[1] = *(const uint4*)&tmp[8];
}

// ---------------- WMMA GEMM + bias + relu ----------------
// Block: 256 threads = 8 waves; wave w computes the 16x16 tile (rows of this
// block) x (columns w*16..w*16+15). K=256 -> 8 x V_WMMA_F32_16X16X32_BF16.
//
// The 8 KB of A fragments for this block are shared by all 8 waves, so stage
// them into LDS once with GLOBAL_LOAD_ASYNC_TO_LDS_B128 (ASYNCcnt), then each
// wave reads its operands with ds_load_b128. B (weights, 64 KB total) stays in
// global: it is reused by all 6250 blocks and lives in WGP$/L2.
__global__ void gemm_relu_kernel(const unsigned short* __restrict__ xafrag,
                                 const unsigned short* __restrict__ wfrag,
                                 const float* __restrict__ bias,
                                 float* __restrict__ out) {
    __shared__ __align__(32) unsigned short aLds[4096];   // 8 KB: 8 ktiles x 32 lanes x 16 bf16

    int rowblk = blockIdx.x;
    int wv   = threadIdx.x >> 5;        // N tile 0..7
    int lane = threadIdx.x & 31;
    int t    = threadIdx.x;

    // Async stage A fragments: 8192 bytes, 256 threads x 2 x 16B.
    {
        const char* g = (const char*)(xafrag + (long)rowblk * 4096);
        char* l = (char*)aLds;
        __builtin_amdgcn_global_load_async_to_lds_b128(
            (as1_v4i*)(g + t * 16),        (as3_v4i*)(l + t * 16),        0, 0);
        __builtin_amdgcn_global_load_async_to_lds_b128(
            (as1_v4i*)(g + 4096 + t * 16), (as3_v4i*)(l + 4096 + t * 16), 0, 0);
    }
    __builtin_amdgcn_s_wait_asynccnt(0);   // my wave's async copies landed in LDS
    __syncthreads();                        // everyone's copies visible

    const v16bf* aL    = (const v16bf*)aLds;                          // ds_load_b128 x2 each
    const v16bf* bBase = (const v16bf*)(wfrag + (long)wv * 8 * 32 * 16);

    v8f acc = {};
#pragma unroll
    for (int kt = 0; kt < 8; ++kt) {
        v16bf a = aL[kt * 32 + lane];
        v16bf b = bBase[kt * 32 + lane];
        acc = __builtin_amdgcn_wmma_f32_16x16x32_bf16(
            /*neg_a=*/false, a, /*neg_b=*/false, b,
            /*c_mod=*/(short)0, acc, /*reuse_a=*/false, /*reuse_b=*/false);
    }

    // C/D layout: lanes 0-15 -> N=lane, M=j (vgpr j); lanes 16-31 -> M=8+j.
    int n = wv * 16 + (lane & 15);
    int mbase = rowblk * 16 + ((lane >> 4) << 3);
    float bv = bias[n];
#pragma unroll
    for (int j = 0; j < 8; ++j) {
        float v = acc[j] + bv;
        out[(long)(mbase + j) * DIM + n] = v > 0.0f ? v : 0.0f;
    }
}

// ---------------- segment max pool (relu output >= 0, hg zero-inited) -----
__global__ void segmax_kernel(const float* __restrict__ h, const int* __restrict__ gid,
                              unsigned int* __restrict__ hg) {
    long idx = (long)blockIdx.x * blockDim.x + threadIdx.x;
    if (idx < (long)N_NODES * DIM) {
        int i = (int)(idx >> 7);
        int f = (int)(idx & 127);
        atomicMax(&hg[(long)gid[i] * DIM + f], __float_as_uint(h[idx]));
    }
}

// ---------------- classifier: out = hg @ Wc + bc (50x10) ----------------
__global__ void classify_kernel(const float* __restrict__ hg, const float* __restrict__ Wc,
                                const float* __restrict__ bc, float* __restrict__ out) {
    int t = blockIdx.x * blockDim.x + threadIdx.x;
    if (t < N_GRAPHS * NCLS) {
        int b = t / NCLS;
        int c = t % NCLS;
        float s = bc[c];
#pragma unroll 4
        for (int f = 0; f < DIM; ++f) s += hg[(long)b * DIM + f] * Wc[(long)f * NCLS + c];
        out[t] = s;
    }
}

// ---------------- workspace offsets (bytes) ----------------
#define OFF_DINV  (0L)                  // N floats            (0.4 MB)
#define OFF_AGG   (1L << 20)            // N*128 floats        (51.2 MB)
#define OFF_XA    (56L << 20)           // N*256 bf16          (51.2 MB)
#define OFF_H     (112L << 20)          // N*128 floats        (51.2 MB)
#define OFF_W1    (168L << 20)          // 256*128 bf16        (64 KB)
#define OFF_W2    (169L << 20)          // 256*128 bf16        (64 KB)
#define OFF_HG    (170L << 20)          // 50*128 floats       (25.6 KB)

extern "C" void kernel_launch(void* const* d_in, const int* in_sizes, int n_in,
                              void* d_out, int out_size, void* d_ws, size_t ws_size,
                              hipStream_t stream) {
    const float* x   = (const float*)d_in[0];
    const float* W1  = (const float*)d_in[1];
    const float* b1  = (const float*)d_in[2];
    const float* W2  = (const float*)d_in[3];
    const float* b2  = (const float*)d_in[4];
    const float* Wc  = (const float*)d_in[5];
    const float* bc  = (const float*)d_in[6];
    const int*   src = (const int*)d_in[7];
    const int*   dst = (const int*)d_in[8];
    const int*   gid = (const int*)d_in[9];
    float* out = (float*)d_out;

    char* ws = (char*)d_ws;
    float*          dinv   = (float*)(ws + OFF_DINV);
    float*          agg    = (float*)(ws + OFF_AGG);
    unsigned short* xafrag = (unsigned short*)(ws + OFF_XA);
    float*          h      = (float*)(ws + OFF_H);
    unsigned short* wfrag1 = (unsigned short*)(ws + OFF_W1);
    unsigned short* wfrag2 = (unsigned short*)(ws + OFF_W2);
    float*          hg     = (float*)(ws + OFF_HG);

    const int T = 256;
    const int SPMM_BLOCKS = N_EDGES / 2;                  // E*128/256 = 800000

    // 0) zero accumulators (every launch: deterministic replay)
    zero_f32<<<dim3((N_NODES + T - 1) / T), T, 0, stream>>>(dinv, N_NODES);
    zero_f32<<<dim3(8192), T, 0, stream>>>(agg, (long)N_NODES * DIM);
    zero_f32<<<dim3((N_GRAPHS * DIM + T - 1) / T), T, 0, stream>>>(hg, (long)N_GRAPHS * DIM);

    // 1) degree -> dinv (in place)
    degree_kernel<<<dim3((N_EDGES + T - 1) / T), T, 0, stream>>>(dst, dinv);
    dinv_kernel<<<dim3((N_NODES + T - 1) / T), T, 0, stream>>>(dinv);

    // 2) weight fragment packs (bf16)
    pack_w_kernel<<<dim3(8), T, 0, stream>>>(W1, wfrag1);
    pack_w_kernel<<<dim3(8), T, 0, stream>>>(W2, wfrag2);

    // 3) layer 1: spmm(x) -> pack -> wmma gemm + relu -> h
    spmm_kernel<<<dim3(SPMM_BLOCKS), T, 0, stream>>>(src, dst, x, dinv, agg);
    pack_a_kernel<<<dim3(ROWBLKS), T, 0, stream>>>(x, agg, dinv, xafrag);
    gemm_relu_kernel<<<dim3(ROWBLKS), T, 0, stream>>>(xafrag, wfrag1, b1, h);

    // 4) layer 2: spmm(h) -> pack -> wmma gemm + relu -> h (in place of h1)
    spmm_kernel<<<dim3(SPMM_BLOCKS), T, 0, stream>>>(src, dst, h, dinv, agg);
    pack_a_kernel<<<dim3(ROWBLKS), T, 0, stream>>>(h, agg, dinv, xafrag);
    gemm_relu_kernel<<<dim3(ROWBLKS), T, 0, stream>>>(xafrag, wfrag2, b2, h);

    // 5) graph max pool + classifier
    segmax_kernel<<<dim3((int)(((long)N_NODES * DIM + T - 1) / T)), T, 0, stream>>>(h, gid, (unsigned int*)hg);
    classify_kernel<<<dim3(2), T, 0, stream>>>(hg, Wc, bc, out);
}
